// QComplexAttention_1752346657401
// MI455X (gfx1250) — matrix operand order repaired and checked by reading, not compiled, not run on hardware
//
#include <hip/hip_runtime.h>
#include <hip/hip_bf16.h>
#include <math.h>

// Problem constants (from reference)
#define B_  2
#define S_  1024
#define E_  1024
#define H_  16
#define DH  64

typedef __attribute__((ext_vector_type(16))) __bf16 v16bf;
typedef __attribute__((ext_vector_type(8)))  __bf16 v8bf;
typedef __attribute__((ext_vector_type(8)))  float  v8f;

static __device__ __forceinline__ v8f wmma_bf16(v16bf a, v16bf b, v8f c) {
  // D = A(16x32 bf16) x B(32x16 bf16) + C(16x16 f32)
  return __builtin_amdgcn_wmma_f32_16x16x32_bf16(false, a, false, b, (short)0, c, false, false);
}

// A fragment (16x32, 16-bit): lane L row = L%16; elems 0..7 -> K = half8 + i,
// elems 8..15 -> K = 16 + half8 + (i-8).  base = &X[row*ldk + k0]
static __device__ __forceinline__ v16bf frag_a(const __bf16* base, int half8) {
  v8bf lo = *(const v8bf*)(base + half8);
  v8bf hi = *(const v8bf*)(base + half8 + 16);
  v16bf r;
#pragma unroll
  for (int i = 0; i < 8; ++i) { r[i] = lo[i]; r[i + 8] = hi[i]; }
  return r;
}

// B fragment (32x16, 16-bit): lane L col = L%16; elems 0..7 -> K = half16/2 + i,
// elems 8..15 -> K = 16 + half16/2 + (i-8).
// base = &W[n*ldk + k0]  (contraction contiguous)
static __device__ __forceinline__ v16bf frag_b(const __bf16* base, int half16) {
  v8bf lo = *(const v8bf*)(base + (half16 >> 1));
  v8bf hi = *(const v8bf*)(base + (half16 >> 1) + 16);
  v16bf r;
#pragma unroll
  for (int i = 0; i < 8; ++i) { r[i] = lo[i]; r[i + 8] = hi[i]; }
  return r;
}

// ---------------------------------------------------------------- conversions
__global__ void cvt_f32_bf16(const float* __restrict__ src,
                             __bf16* __restrict__ dst, int n) {
  int i = blockIdx.x * blockDim.x + threadIdx.x;
  if (i < n) dst[i] = (__bf16)src[i];
}

// RoPE + (B,S,E)->(B,H,S,Dh) reshape + bf16.  One thread per (b,s,h,j), j<Dh/2.
__global__ void rope_shape_bf16(const float* __restrict__ src,
                                __bf16* __restrict__ dst) {
  int tid = blockIdx.x * blockDim.x + threadIdx.x;   // B*S*H*32 = 2^20
  int j = tid & 31;
  int h = (tid >> 5) & (H_ - 1);
  int s = (tid >> 9) & (S_ - 1);
  int b = tid >> 19;
  size_t si = ((size_t)(b * S_ + s)) * E_ + h * DH + j;
  float x1 = src[si];
  float x2 = src[si + 32];
  float ang = (float)s * powf(10000.0f, -(float)j * (1.0f / 32.0f));
  float sn, c;
  sincosf(ang, &sn, &c);
  size_t o = ((size_t)((b * H_ + h) * S_ + s)) * DH + j;
  dst[o]      = (__bf16)(x1 * c - x2 * sn);
  dst[o + 32] = (__bf16)(x2 * c + x1 * sn);
}

// (B,S,E) f32 -> transposed (B,H,Dh,S) bf16 for V (contraction contiguous)
__global__ void vshape_bf16(const float* __restrict__ src,
                            __bf16* __restrict__ dst) {
  int tid = blockIdx.x * blockDim.x + threadIdx.x;   // B*S*E = 2^21
  int d = tid & (DH - 1);
  int h = (tid >> 6) & (H_ - 1);
  int s = (tid >> 10) & (S_ - 1);
  int b = tid >> 20;
  dst[((size_t)((b * H_ + h) * DH + d)) * S_ + s] = (__bf16)src[tid];
}

// ------------------------------------------------ dual (quaternion) dense GEMM
// out_a = xa@wa.T + j2*(xb@wb.T) + ba ;  out_b = xa@wb.T + xb@wa.T + bb
// One wave computes a 32x32 output macro-tile (2x2 register blocking):
// per K=32 step: 8 fragment loads (16 x b128) feed 16 WMMAs.
__global__ __launch_bounds__(256) void qc_dense_wmma(
    const __bf16* __restrict__ xa, const __bf16* __restrict__ xb,
    const __bf16* __restrict__ wa, const __bf16* __restrict__ wb,
    const float* __restrict__ ba, const float* __restrict__ bb,
    const float* __restrict__ theta_p,
    float* __restrict__ outa, float* __restrict__ outb,
    int M, int N, int K) {
  int lane = threadIdx.x & 31;
  int wid  = (blockIdx.x * blockDim.x + threadIdx.x) >> 5;
  int nmac = N >> 5;                    // macro-tiles along N
  int mt0 = (wid / nmac) * 2;           // first 16-row tile of the 32x32 block
  int nt0 = (wid % nmac) * 2;
  int mn = lane & 15;
  int half8  = (lane & 16) ? 8  : 0;
  int half16 = (lane & 16) ? 16 : 0;

  const __bf16* pa[2];  const __bf16* pb[2];
  const __bf16* pwa[2]; const __bf16* pwb[2];
#pragma unroll
  for (int i = 0; i < 2; ++i) {
    pa[i]  = xa + (size_t)((mt0 + i) * 16 + mn) * K;
    pb[i]  = xb + (size_t)((mt0 + i) * 16 + mn) * K;
    pwa[i] = wa + (size_t)((nt0 + i) * 16 + mn) * K;
    pwb[i] = wb + (size_t)((nt0 + i) * 16 + mn) * K;
  }

  v8f acc_aa[2][2], acc_ab[2][2], acc_b[2][2];
#pragma unroll
  for (int mi = 0; mi < 2; ++mi)
#pragma unroll
    for (int ni = 0; ni < 2; ++ni) {
      acc_aa[mi][ni] = (v8f){};
      acc_ab[mi][ni] = (v8f){};
      acc_b[mi][ni]  = (v8f){};
    }

  for (int k = 0; k < K; k += 32) {
    v16bf axa[2], axb[2], bwa[2], bwb[2];
#pragma unroll
    for (int i = 0; i < 2; ++i) {
      axa[i] = frag_a(pa[i] + k, half8);
      axb[i] = frag_a(pb[i] + k, half8);
      bwa[i] = frag_b(pwa[i] + k, half16);
      bwb[i] = frag_b(pwb[i] + k, half16);
    }
#pragma unroll
    for (int mi = 0; mi < 2; ++mi)
#pragma unroll
      for (int ni = 0; ni < 2; ++ni) {
        acc_aa[mi][ni] = wmma_bf16(axa[mi], bwa[ni], acc_aa[mi][ni]); // xa.wa^T
        acc_ab[mi][ni] = wmma_bf16(axb[mi], bwb[ni], acc_ab[mi][ni]); // xb.wb^T
        acc_b[mi][ni]  = wmma_bf16(axa[mi], bwb[ni], acc_b[mi][ni]);  // xa.wb^T
        acc_b[mi][ni]  = wmma_bf16(axb[mi], bwa[ni], acc_b[mi][ni]);  // xb.wa^T
      }
  }

  float j2 = sinf(2.0f * theta_p[0]) - 1.0f;
#pragma unroll
  for (int ni = 0; ni < 2; ++ni) {
    int n = (nt0 + ni) * 16 + mn;
    float bia = ba[n], bib = bb[n];
#pragma unroll
    for (int mi = 0; mi < 2; ++mi)
#pragma unroll
      for (int r = 0; r < 8; ++r) {     // C layout: row = r + half8, col = n
        int row = (mt0 + mi) * 16 + r + half8;
        outa[(size_t)row * N + n] =
            acc_aa[mi][ni][r] + j2 * acc_ab[mi][ni][r] + bia;
        outb[(size_t)row * N + n] = acc_b[mi][ni][r] + bib;
      }
  }
}

// ------------------------------------------------------- flash QC attention
// One wave per (b, h, 16 query rows).  Key tiles processed in pairs so the
// softmax P block is a full 16x32 bf16 WMMA A operand.
__global__ __launch_bounds__(256) void flash_qc(
    const __bf16* __restrict__ qa, const __bf16* __restrict__ qb,
    const __bf16* __restrict__ ka, const __bf16* __restrict__ kb,
    const __bf16* __restrict__ vta, const __bf16* __restrict__ vtb,
    const float* __restrict__ thetas_head,
    float* __restrict__ oa_out, float* __restrict__ ob_out) {
  __shared__ __align__(16) __bf16 Pl[8][16][32];   // per-wave P staging (1KB/wave)
  int lane = threadIdx.x & 31;
  int w = threadIdx.x >> 5;
  int gw = blockIdx.x * 8 + w;          // 0..2047
  int qtile = gw & 63;                  // S/16
  int bh = gw >> 6;                     // b*H + h
  int b = bh >> 4, h = bh & 15;
  int mn = lane & 15;
  int half8  = (lane & 16) ? 8  : 0;
  int half16 = (lane & 16) ? 16 : 0;
  float j2h = sinf(2.0f * thetas_head[h]) - 1.0f;

  const __bf16* qa_p = qa + ((size_t)bh * S_ + qtile * 16 + mn) * DH;
  const __bf16* qb_p = qb + ((size_t)bh * S_ + qtile * 16 + mn) * DH;
  v16bf qaf[2], qbf[2];
#pragma unroll
  for (int kk = 0; kk < 2; ++kk) {      // Dh=64 -> two K=32 steps
    qaf[kk] = frag_a(qa_p + kk * 32, half8);
    qbf[kk] = frag_a(qb_p + kk * 32, half8);
  }

  v8f oa[4] = {}, ob[4] = {};
  float mrow[8], lrow[8];
#pragma unroll
  for (int r = 0; r < 8; ++r) { mrow[r] = -INFINITY; lrow[r] = 0.0f; }

  int npairs = (qtile >> 1) + 1;
  for (int pp = 0; pp < npairs; ++pp) {
    int j0 = pp * 2;
    v8f mag[2];
#pragma unroll
    for (int t2 = 0; t2 < 2; ++t2) {
      int jt = j0 + t2;                 // wave-uniform
      if (jt > qtile) {                 // fully masked tile of the pair
#pragma unroll
        for (int r = 0; r < 8; ++r) mag[t2][r] = -1e9f;
        continue;
      }
      const __bf16* ka_p = ka + ((size_t)bh * S_ + jt * 16 + mn) * DH;
      const __bf16* kb_p = kb + ((size_t)bh * S_ + jt * 16 + mn) * DH;
      v16bf kaf0 = frag_b(ka_p,      half16);
      v16bf kaf1 = frag_b(ka_p + 32, half16);
      v16bf kbf0 = frag_b(kb_p,      half16);
      v16bf kbf1 = frag_b(kb_p + 32, half16);
      v8f z = {};
      v8f sa = wmma_bf16(qaf[0], kaf0, z);  sa = wmma_bf16(qaf[1], kaf1, sa);
      v8f tb = wmma_bf16(qbf[0], kbf0, z);  tb = wmma_bf16(qbf[1], kbf1, tb);
      v8f sb = wmma_bf16(qaf[0], kbf0, z);  sb = wmma_bf16(qaf[1], kbf1, sb);
      sb = wmma_bf16(qbf[0], kaf0, sb);     sb = wmma_bf16(qbf[1], kaf1, sb);
#pragma unroll
      for (int r = 0; r < 8; ++r) {
        float A  = sa[r] + j2h * tb[r];
        float Bc = sb[r];
        float mg = sqrtf(A * A + Bc * Bc + 1e-8f) * 0.125f;  // /sqrt(Dh)
        if (jt == qtile) {              // diagonal tile: causal mask
          int col = jt * 16 + mn;
          int row = qtile * 16 + r + half8;
          if (col > row) mg = -1e9f;
        }
        mag[t2][r] = mg;
      }
    }
    // online softmax over the 32 new columns; row reductions across the 16
    // lanes of each wave32 half (xor 1/2/4/8 stays inside the half).
#pragma unroll
    for (int r = 0; r < 8; ++r) {
      float mx = fmaxf(mag[0][r], mag[1][r]);
      mx = fmaxf(mx, __shfl_xor(mx, 1, 32));
      mx = fmaxf(mx, __shfl_xor(mx, 2, 32));
      mx = fmaxf(mx, __shfl_xor(mx, 4, 32));
      mx = fmaxf(mx, __shfl_xor(mx, 8, 32));
      float mnew  = fmaxf(mrow[r], mx);
      float scale = expf(mrow[r] - mnew);   // expf(-inf)=0 on first pair
      float p0 = expf(mag[0][r] - mnew);
      float p1 = expf(mag[1][r] - mnew);
      float rs = p0 + p1;
      rs += __shfl_xor(rs, 1, 32);
      rs += __shfl_xor(rs, 2, 32);
      rs += __shfl_xor(rs, 4, 32);
      rs += __shfl_xor(rs, 8, 32);
      lrow[r] = lrow[r] * scale + rs;
      mrow[r] = mnew;
#pragma unroll
      for (int t = 0; t < 4; ++t) { oa[t][r] *= scale; ob[t][r] *= scale; }
      Pl[w][r + half8][mn]      = (__bf16)p0;   // C layout -> row-major 16x32
      Pl[w][r + half8][16 + mn] = (__bf16)p1;
    }
    // Re-load P as an A fragment (wave-local LDS; DS ops are in-order/wave).
    v16bf pf;
    {
      const __bf16* pbase = &Pl[w][mn][0];
      v8bf lo = *(const v8bf*)(pbase + half8);
      v8bf hi = *(const v8bf*)(pbase + half8 + 16);
#pragma unroll
      for (int i = 0; i < 8; ++i) { pf[i] = lo[i]; pf[i + 8] = hi[i]; }
    }
    // O += P x V over the 32 key rows; V is pre-transposed (B,H,Dh,S) so the
    // B fragment is two contiguous b128 loads.
#pragma unroll
    for (int t = 0; t < 4; ++t) {
      const __bf16* va_p = vta + ((size_t)bh * DH + t * 16 + mn) * S_ + j0 * 16;
      const __bf16* vb_p = vtb + ((size_t)bh * DH + t * 16 + mn) * S_ + j0 * 16;
      v16bf vaf = frag_b(va_p, half16);
      v16bf vbf = frag_b(vb_p, half16);
      oa[t] = wmma_bf16(pf, vaf, oa[t]);
      ob[t] = wmma_bf16(pf, vbf, ob[t]);
    }
  }
  // Normalize and un-shape directly to (B,S,E)
#pragma unroll
  for (int r = 0; r < 8; ++r) {
    float inv = 1.0f / lrow[r];
    int row = qtile * 16 + r + half8;
#pragma unroll
    for (int t = 0; t < 4; ++t) {
      int d = h * DH + t * 16 + mn;
      oa_out[((size_t)b * S_ + row) * E_ + d] = oa[t][r] * inv;
      ob_out[((size_t)b * S_ + row) * E_ + d] = ob[t][r] * inv;
    }
  }
}

// ---------------------------------------------------------------- launcher
extern "C" void kernel_launch(void* const* d_in, const int* in_sizes, int n_in,
                              void* d_out, int out_size, void* d_ws, size_t ws_size,
                              hipStream_t stream) {
  (void)in_sizes; (void)n_in; (void)out_size; (void)ws_size;
  const int BSE = B_ * S_ * E_;   // 2,097,152
  const int EE  = E_ * E_;        // 1,048,576
  const size_t MB = 1u << 20;
  char* W = (char*)d_ws;

  // bf16 activations (0..16MB)
  __bf16* xqa  = (__bf16*)(W + 0 * MB);
  __bf16* xqb  = (__bf16*)(W + 4 * MB);
  __bf16* xkva = (__bf16*)(W + 8 * MB);
  __bf16* xkvb = (__bf16*)(W + 12 * MB);
  // bf16 weights (16..32MB)
  __bf16* wqa = (__bf16*)(W + 16 * MB);
  __bf16* wqb = (__bf16*)(W + 18 * MB);
  __bf16* wka = (__bf16*)(W + 20 * MB);
  __bf16* wkb = (__bf16*)(W + 22 * MB);
  __bf16* wva = (__bf16*)(W + 24 * MB);
  __bf16* wvb = (__bf16*)(W + 26 * MB);
  __bf16* woa = (__bf16*)(W + 28 * MB);
  __bf16* wob = (__bf16*)(W + 30 * MB);
  // dense f32 outputs (32..80MB)
  float* qa_f = (float*)(W + 32 * MB);
  float* qb_f = (float*)(W + 40 * MB);
  float* ka_f = (float*)(W + 48 * MB);
  float* kb_f = (float*)(W + 56 * MB);
  float* va_f = (float*)(W + 64 * MB);
  float* vb_f = (float*)(W + 72 * MB);
  // shaped bf16 for attention (80..104MB)
  __bf16* qa_s = (__bf16*)(W + 80 * MB);
  __bf16* qb_s = (__bf16*)(W + 84 * MB);
  __bf16* ka_s = (__bf16*)(W + 88 * MB);
  __bf16* kb_s = (__bf16*)(W + 92 * MB);
  __bf16* vta  = (__bf16*)(W + 96 * MB);
  __bf16* vtb  = (__bf16*)(W + 100 * MB);
  // attention outputs (reuse 32..56MB; dense f32 is dead by then)
  float*  aoa    = (float*)(W + 32 * MB);
  float*  aob    = (float*)(W + 40 * MB);
  __bf16* aoa_bf = (__bf16*)(W + 48 * MB);
  __bf16* aob_bf = (__bf16*)(W + 52 * MB);

  const float* theta  = (const float*)d_in[5];
  const float* thetas = (const float*)d_in[6];

  // 1. convert activations & weights to bf16
  cvt_f32_bf16<<<BSE / 256, 256, 0, stream>>>((const float*)d_in[0], xqa,  BSE);
  cvt_f32_bf16<<<BSE / 256, 256, 0, stream>>>((const float*)d_in[1], xqb,  BSE);
  cvt_f32_bf16<<<BSE / 256, 256, 0, stream>>>((const float*)d_in[2], xkva, BSE);
  cvt_f32_bf16<<<BSE / 256, 256, 0, stream>>>((const float*)d_in[3], xkvb, BSE);
  cvt_f32_bf16<<<EE / 256, 256, 0, stream>>>((const float*)d_in[7],  wqa, EE);
  cvt_f32_bf16<<<EE / 256, 256, 0, stream>>>((const float*)d_in[8],  wqb, EE);
  cvt_f32_bf16<<<EE / 256, 256, 0, stream>>>((const float*)d_in[11], wka, EE);
  cvt_f32_bf16<<<EE / 256, 256, 0, stream>>>((const float*)d_in[12], wkb, EE);
  cvt_f32_bf16<<<EE / 256, 256, 0, stream>>>((const float*)d_in[15], wva, EE);
  cvt_f32_bf16<<<EE / 256, 256, 0, stream>>>((const float*)d_in[16], wvb, EE);
  cvt_f32_bf16<<<EE / 256, 256, 0, stream>>>((const float*)d_in[19], woa, EE);
  cvt_f32_bf16<<<EE / 256, 256, 0, stream>>>((const float*)d_in[20], wob, EE);

  // 2. q/k/v dual dense projections (M=2048,N=K=1024):
  //    (M/32)*(N/32) = 2048 waves -> 256 blocks of 8 waves
  const int dense_blocks = ((B_ * S_ / 32) * (E_ / 32)) / 8;   // 256
  qc_dense_wmma<<<dense_blocks, 256, 0, stream>>>(
      xqa, xqb, wqa, wqb, (const float*)d_in[9], (const float*)d_in[10],
      theta, qa_f, qb_f, B_ * S_, E_, E_);
  qc_dense_wmma<<<dense_blocks, 256, 0, stream>>>(
      xkva, xkvb, wka, wkb, (const float*)d_in[13], (const float*)d_in[14],
      theta, ka_f, kb_f, B_ * S_, E_, E_);
  qc_dense_wmma<<<dense_blocks, 256, 0, stream>>>(
      xkva, xkvb, wva, wvb, (const float*)d_in[17], (const float*)d_in[18],
      theta, va_f, vb_f, B_ * S_, E_, E_);

  // 3. RoPE+shape for q/k, transpose-shape for v
  const int rope_blocks = (B_ * S_ * H_ * 32) / 256;   // 4096
  rope_shape_bf16<<<rope_blocks, 256, 0, stream>>>(qa_f, qa_s);
  rope_shape_bf16<<<rope_blocks, 256, 0, stream>>>(qb_f, qb_s);
  rope_shape_bf16<<<rope_blocks, 256, 0, stream>>>(ka_f, ka_s);
  rope_shape_bf16<<<rope_blocks, 256, 0, stream>>>(kb_f, kb_s);
  vshape_bf16<<<BSE / 256, 256, 0, stream>>>(va_f, vta);
  vshape_bf16<<<BSE / 256, 256, 0, stream>>>(vb_f, vtb);

  // 4. flash attention: 2048 waves (B*H*64 q-tiles) / 8 waves per block
  flash_qc<<<256, 256, 0, stream>>>(qa_s, qb_s, ka_s, kb_s, vta, vtb, thetas,
                                    aoa, aob);

  // 5. output projection (writes final tuple directly to d_out)
  cvt_f32_bf16<<<BSE / 256, 256, 0, stream>>>(aoa, aoa_bf, BSE);
  cvt_f32_bf16<<<BSE / 256, 256, 0, stream>>>(aob, aob_bf, BSE);
  float* outA = (float*)d_out;
  float* outB = outA + BSE;
  qc_dense_wmma<<<dense_blocks, 256, 0, stream>>>(
      aoa_bf, aob_bf, woa, wob, (const float*)d_in[21], (const float*)d_in[22],
      theta, outA, outB, B_ * S_, E_, E_);
}
